// SelfAttentionHead_33389075759139
// MI455X (gfx1250) — matrix operand order repaired and checked by reading, not compiled, not run on hardware
//
#include <hip/hip_runtime.h>
#include <hip/hip_bf16.h>

// ---------------------------------------------------------------------------
// Self-attention head, MI455X (gfx1250), wave32, WMMA bf16 16x16x32.
//   x:[8,2048,2048]f32, Wq/Wk/Wv:[128,2048]f32 -> out:[8,2048,128]f32
// Phase 0: repack weights f32->bf16 into WMMA B-fragment order.
// Phase 1: QKV projection (double-buffered x tile in LDS, ping-ponged
//          B-fragment prefetch so WMMAs never wait on L2).
// Phase 2: flash attention (double-buffered K/V^T tiles, async-to-LDS DMA,
//          online softmax, all matmuls via v_wmma).
// Workspace: Wfrag@0 (1.5MB), Q@2MB (4MB), K@6MB (4MB), V^T@10MB (4MB).
// ---------------------------------------------------------------------------

#define EMBED 2048
#define HEAD  128
#define SEQ   2048
#define BATCH 8

typedef __attribute__((ext_vector_type(16))) __bf16 v16bf;
typedef __attribute__((ext_vector_type(8)))  __bf16 v8bf;
typedef __attribute__((ext_vector_type(8)))  float  v8f;
typedef int v4i __attribute__((vector_size(16)));    // async builtin pointee

union FragU { v16bf v; uint4 q[2]; };

__device__ __forceinline__ v8f zero_v8f() {
    v8f z;
    #pragma unroll
    for (int i = 0; i < 8; ++i) z[i] = 0.0f;
    return z;
}

// --- CDNA5 async global->LDS path (guarded so fallback still compiles) -----
#if defined(__has_builtin)
#  if __has_builtin(__builtin_amdgcn_global_load_async_to_lds_b128) && \
      __has_builtin(__builtin_amdgcn_s_wait_asynccnt)
#    define USE_ASYNC_LDS 1
#  endif
#endif
#ifndef USE_ASYNC_LDS
#  define USE_ASYNC_LDS 0
#endif

#if USE_ASYNC_LDS
#define ASYNC_CP16(gsrc, ldst)                                                \
    __builtin_amdgcn_global_load_async_to_lds_b128(                           \
        (__attribute__((address_space(1))) v4i*)(gsrc),                       \
        (__attribute__((address_space(3))) v4i*)(ldst), 0, 0)
#define WAIT_ASYNC(n) __builtin_amdgcn_s_wait_asynccnt(n)
#endif

// ---------------------------------------------------------------------------
// Kernel 0: repack W[128][2048] f32 -> bf16 B-fragments.
// Fragment id = (mat*8 + ntile)*64 + kchunk ; 512 bf16 per fragment.
// B-frag layout (16-bit, 32x16 KxN): lane l holds column n=l%16,
// K-values 16*(l/16) .. +15 (contiguous along C in W's row-major storage).
// ---------------------------------------------------------------------------
__global__ void repack_w_kernel(const float* __restrict__ Wq,
                                const float* __restrict__ Wk,
                                const float* __restrict__ Wv,
                                __bf16* __restrict__ wf) {
    const int frag = blockIdx.x;          // 0..1535
    const int kc  = frag & 63;
    const int nt  = (frag >> 6) & 7;
    const int mat = frag >> 9;
    const float* W = (mat == 0) ? Wq : ((mat == 1) ? Wk : Wv);
    const int l  = threadIdx.x;           // 0..31 (one wave)
    const int h  = nt * 16 + (l & 15);
    const int c0 = kc * 32 + (l >> 4) * 16;
    const float* src = W + (size_t)h * EMBED + c0;
    __bf16* dst = wf + (size_t)frag * 512 + l * 16;
    v8bf p0, p1;
    #pragma unroll
    for (int i = 0; i < 8; ++i) {
        p0[i] = (__bf16)src[i];
        p1[i] = (__bf16)src[8 + i];
    }
    *(v8bf*)(dst)     = p0;
    *(v8bf*)(dst + 8) = p1;
}

// ---------------------------------------------------------------------------
// Kernel 1: QKV projection. Block = 128 threads (4 waves), 16 flat rows.
// Wave w owns n-tiles {2w, 2w+1} of all three outputs (6 f32 accumulators).
// Double-buffered x tile in LDS; B-fragments ping-ponged one iteration ahead.
// ---------------------------------------------------------------------------
__global__ __launch_bounds__(128) void proj_qkv_kernel(
        const float*  __restrict__ x,
        const __bf16* __restrict__ wf,
        __bf16* __restrict__ Qo,      // [B*T][128] row-major
        __bf16* __restrict__ Ko,      // [B*T][128] row-major
        __bf16* __restrict__ Vt) {    // [B][128][T] transposed
    __shared__ __align__(16) float xs[2][16 * 32];

    const int rb  = blockIdx.x * 16;       // flat row base over B*T
    const int tid = threadIdx.x;
    const int w   = tid >> 5;              // wave 0..3
    const int l   = tid & 31;
    const int n   = l & 15;
    const int g   = l >> 4;

    // cooperative stager: 128 threads x 16B per tile
    const int srow = tid >> 3;
    const int scol = (tid & 7) * 4;
    const float* xg = x + (size_t)(rb + srow) * EMBED + scol;

    v8f acc[3][2];
    #pragma unroll
    for (int m = 0; m < 3; ++m)
        #pragma unroll
        for (int j = 0; j < 2; ++j) acc[m][j] = zero_v8f();

    // per-wave fragment base: wf + ((m*8 + 2w+j)*64 + kc)*512 + l*16
    const __bf16* wfb = wf + ((size_t)(w * 2) * 64) * 512 + l * 16;

    auto loadB = [&](int kc, FragU (&B)[6]) {
        #pragma unroll
        for (int m = 0; m < 3; ++m)
            #pragma unroll
            for (int j = 0; j < 2; ++j) {
                const uint4* bp = (const uint4*)(wfb
                    + ((size_t)m * 8 * 64 + (size_t)j * 64 + kc) * 512);
                B[m * 2 + j].q[0] = bp[0];
                B[m * 2 + j].q[1] = bp[1];
            }
    };

    // one pipelined k-iteration: uses Bu (already in flight), prefetches Bp
    auto step = [&](int kc, FragU (&Bu)[6], FragU (&Bp)[6]) {
        if (kc + 1 < 64) loadB(kc + 1, Bp);          // next frags in flight
        float4 nxt;
        if (kc < 63) nxt = *(const float4*)(xg + (size_t)(kc + 1) * 32);
        __syncthreads();

        const int cur = kc & 1;
        // A-fragment (16x32 bf16): lane holds row m=l%16,
        // K = {16g..16g+15}
        FragU A;
        {
            const int kb = g * 16;
            #pragma unroll
            for (int i = 0; i < 8; ++i) {
                A.v[i]     = (__bf16)xs[cur][n * 32 + kb + i];
                A.v[8 + i] = (__bf16)xs[cur][n * 32 + 8 + kb + i];
            }
        }
        #pragma unroll
        for (int m = 0; m < 3; ++m)
            #pragma unroll
            for (int j = 0; j < 2; ++j)
                acc[m][j] = __builtin_amdgcn_wmma_f32_16x16x32_bf16(
                    false, A.v, false, Bu[m * 2 + j].v, (short)0,
                    acc[m][j], false, false);

        if (kc < 63) *(float4*)&xs[1 - cur][srow * 32 + scol] = nxt;
    };

    // prologue: stage x tile 0 + B-fragments for kc=0
    *(float4*)&xs[0][srow * 32 + scol] = *(const float4*)(xg);
    FragU B0[6], B1[6];
    loadB(0, B0);

    for (int kc = 0; kc < 64; kc += 2) {
        step(kc,     B0, B1);
        step(kc + 1, B1, B0);
    }

    // Epilogue. C/D layout: lane l, vgpr r -> [m = r + 8g][n = l%16].
    const int b = rb / SEQ;
    const int t = (rb % SEQ) + g * 8;
    #pragma unroll
    for (int j = 0; j < 2; ++j) {
        const int nt = w * 2 + j;
        const int h  = nt * 16 + n;
        #pragma unroll
        for (int r = 0; r < 8; ++r) {
            const size_t row = (size_t)rb + g * 8 + r;
            Qo[row * HEAD + h] = (__bf16)acc[0][j][r];
            Ko[row * HEAD + h] = (__bf16)acc[1][j][r];
        }
        // V^T: 8 consecutive t's at fixed h -> one 16B store
        v8bf pk;
        #pragma unroll
        for (int r = 0; r < 8; ++r) pk[r] = (__bf16)acc[2][j][r];
        *(v8bf*)(Vt + (size_t)b * HEAD * SEQ + (size_t)h * SEQ + t) = pk;
    }
}

// ---------------------------------------------------------------------------
// Kernel 2: flash attention. Grid (T/64, B), block 128 (4 waves).
// Wave w: 16 query rows. 32-key tiles, double-buffered in LDS
// (async global->LDS DMA), online softmax.
// ---------------------------------------------------------------------------
__global__ __launch_bounds__(128) void flash_attn_kernel(
        const __bf16* __restrict__ Qo,
        const __bf16* __restrict__ Ko,
        const __bf16* __restrict__ Vt,
        float* __restrict__ out) {
    __shared__ __align__(16) __bf16 Ks[2][32 * HEAD];   // [key][h]   2x8KB
    __shared__ __align__(16) __bf16 Vs[2][HEAD * 32];   // [h][key]   2x8KB
    __shared__ __align__(16) __bf16 Ps[4][16 * 32];     // per-wave P staging

    const int b   = blockIdx.y;
    const int qt  = blockIdx.x;
    const int tid = threadIdx.x;
    const int w   = tid >> 5;
    const int l   = tid & 31;
    const int n   = l & 15;
    const int g   = l >> 4;
    const int q0  = qt * 64 + w * 16;     // wave's query base

    // staging addresses (per thread): K tile row/chunk, V^T tile h-row
    const int krow = tid >> 2;
    const int kco  = (tid & 3) * 32;
    const __bf16* Kg = Ko + ((size_t)(b * SEQ + krow)) * HEAD + kco;
    const __bf16* Vg = Vt + (size_t)b * HEAD * SEQ + (size_t)tid * SEQ;

    // Preload Q A-fragments for the 4 k-chunks of H=128.
    FragU Qf[4];
    {
        const __bf16* qrow = Qo + ((size_t)(b * SEQ + q0 + n)) * HEAD;
        #pragma unroll
        for (int kc = 0; kc < 4; ++kc) {
            Qf[kc].q[0] = *(const uint4*)(qrow + kc * 32 + g * 16);
            Qf[kc].q[1] = *(const uint4*)(qrow + kc * 32 + 8 + g * 16);
        }
    }

    v8f O[8];
    #pragma unroll
    for (int ht = 0; ht < 8; ++ht) O[ht] = zero_v8f();
    float rmax[8], rsum[8];
    #pragma unroll
    for (int r = 0; r < 8; ++r) { rmax[r] = -3.0e38f; rsum[r] = 0.0f; }

    const float scale = 0.022097086912079608f;   // 1/sqrt(2048)
    const int nkt = qt * 2 + 2;                  // causal: keys < q0_block+64

    // --- tile stager: 8 x 16B per thread per tile ---------------------------
    auto stage = [&](int kt, int buf) {
        const size_t k0 = (size_t)kt * 32;
        const __bf16* kg = Kg + k0 * HEAD;
        __bf16*       ks = &Ks[buf][krow * HEAD + kco];
        const __bf16* vg = Vg + k0;
        __bf16*       vs = &Vs[buf][tid * 32];
#if USE_ASYNC_LDS
        #pragma unroll
        for (int i = 0; i < 4; ++i) ASYNC_CP16(kg + i * 8, ks + i * 8);
        #pragma unroll
        for (int i = 0; i < 4; ++i) ASYNC_CP16(vg + i * 8, vs + i * 8);
#else
        const uint4* gk = (const uint4*)kg;
        const uint4* gv = (const uint4*)vg;
        uint4* sk = (uint4*)ks;
        uint4* sv = (uint4*)vs;
        #pragma unroll
        for (int i = 0; i < 4; ++i) sk[i] = gk[i];
        #pragma unroll
        for (int i = 0; i < 4; ++i) sv[i] = gv[i];
#endif
    };

    stage(0, 0);                              // prologue: tile 0 in flight

    for (int kt = 0; kt < nkt; ++kt) {
        const int buf = kt & 1;
        const int k0  = kt * 32;

        __syncthreads();                      // everyone done reading buf
        if (kt + 1 < nkt) {
            stage(kt + 1, buf ^ 1);           // next tile's DMA in flight
#if USE_ASYNC_LDS
            WAIT_ASYNC(8);                    // in-order => tile kt resident
#endif
        } else {
#if USE_ASYNC_LDS
            WAIT_ASYNC(0);
#endif
        }
        __syncthreads();                      // current buf fully written

        // S = Q K^T (16 x 32), two 16-key n-tiles, K-dim = H = 128.
        v8f S[2];
        #pragma unroll
        for (int j = 0; j < 2; ++j) {
            FragU Bf[4];                      // lane: column key=j*16+n,
            #pragma unroll                    // 16 consecutive h per frag
            for (int kc = 0; kc < 4; ++kc) {
                const uint4* bp = (const uint4*)(&Ks[buf][0]
                    + (j * 16 + n) * HEAD + kc * 32 + g * 16);
                Bf[kc].q[0] = bp[0];
                Bf[kc].q[1] = bp[1];
            }
            v8f s = zero_v8f();
            #pragma unroll
            for (int kc = 0; kc < 4; ++kc)
                s = __builtin_amdgcn_wmma_f32_16x16x32_bf16(
                    false, Qf[kc].v, false, Bf[kc].v, (short)0, s, false, false);
            S[j] = s;
        }

        // Scale + causal mask; per-row (over keys) tile max.
        float tmax[8];
        #pragma unroll
        for (int r = 0; r < 8; ++r) {
            const int qrow = q0 + g * 8 + r;
            float s0 = S[0][r] * scale;
            float s1 = S[1][r] * scale;
            if (k0 + n      > qrow) s0 = -3.0e38f;
            if (k0 + 16 + n > qrow) s1 = -3.0e38f;
            S[0][r] = s0; S[1][r] = s1;
            tmax[r] = fmaxf(s0, s1);
        }
        #pragma unroll
        for (int off = 1; off < 16; off <<= 1) {
            #pragma unroll
            for (int r = 0; r < 8; ++r)
                tmax[r] = fmaxf(tmax[r], __shfl_xor(tmax[r], off, 32));
        }

        // Online softmax update.
        float alpha[8];
        #pragma unroll
        for (int r = 0; r < 8; ++r) {
            const float nm = fmaxf(rmax[r], tmax[r]);
            alpha[r] = __expf(rmax[r] - nm);
            rmax[r]  = nm;
        }
        float psum[8];
        #pragma unroll
        for (int r = 0; r < 8; ++r) {
            const float p0 = __expf(S[0][r] - rmax[r]);
            const float p1 = __expf(S[1][r] - rmax[r]);
            S[0][r] = p0; S[1][r] = p1;
            psum[r] = p0 + p1;
        }
        #pragma unroll
        for (int off = 1; off < 16; off <<= 1) {
            #pragma unroll
            for (int r = 0; r < 8; ++r)
                psum[r] += __shfl_xor(psum[r], off, 32);
        }
        #pragma unroll
        for (int r = 0; r < 8; ++r) rsum[r] = rsum[r] * alpha[r] + psum[r];
        #pragma unroll
        for (int ht = 0; ht < 8; ++ht)
            #pragma unroll
            for (int r = 0; r < 8; ++r) O[ht][r] *= alpha[r];

        // Re-layout P: C/D layout -> A-fragment, via per-wave LDS staging.
        __bf16* pb = &Ps[w][0];
        #pragma unroll
        for (int j = 0; j < 2; ++j)
            #pragma unroll
            for (int r = 0; r < 8; ++r)
                pb[(g * 8 + r) * 32 + j * 16 + n] = (__bf16)S[j][r];
        FragU Pf;
        {
            const int kb = g * 16;
            Pf.q[0] = *(const uint4*)(pb + n * 32 + kb);
            Pf.q[1] = *(const uint4*)(pb + n * 32 + 8 + kb);
        }

        // O += P @ V ; B-frag: lane column h=ht*16+n, 16 consecutive keys.
        #pragma unroll
        for (int half = 0; half < 2; ++half) {
            FragU Vf[4];
            #pragma unroll
            for (int i = 0; i < 4; ++i) {
                const int ht = half * 4 + i;
                const uint4* vp = (const uint4*)(&Vs[buf][0]
                    + (ht * 16 + n) * 32 + g * 16);
                Vf[i].q[0] = vp[0];
                Vf[i].q[1] = vp[1];
            }
            #pragma unroll
            for (int i = 0; i < 4; ++i) {
                const int ht = half * 4 + i;
                O[ht] = __builtin_amdgcn_wmma_f32_16x16x32_bf16(
                    false, Pf.v, false, Vf[i].v, (short)0, O[ht], false, false);
            }
        }
    }

    // Finalize: one reciprocal per row, then scaled stores.
    float inv[8];
    #pragma unroll
    for (int r = 0; r < 8; ++r) inv[r] = 1.0f / rsum[r];
    #pragma unroll
    for (int ht = 0; ht < 8; ++ht) {
        const int h = ht * 16 + n;
        #pragma unroll
        for (int r = 0; r < 8; ++r) {
            const int qrow = q0 + g * 8 + r;
            out[((size_t)(b * SEQ + qrow)) * HEAD + h] = O[ht][r] * inv[r];
        }
    }
}

// ---------------------------------------------------------------------------
extern "C" void kernel_launch(void* const* d_in, const int* in_sizes, int n_in,
                              void* d_out, int out_size, void* d_ws, size_t ws_size,
                              hipStream_t stream) {
    const float* x  = (const float*)d_in[0];
    const float* Wq = (const float*)d_in[1];
    const float* Wk = (const float*)d_in[2];
    const float* Wv = (const float*)d_in[3];

    char* ws = (char*)d_ws;
    __bf16* wf = (__bf16*)(ws);                        // 1.5 MB fragments
    __bf16* Q  = (__bf16*)(ws + ((size_t)2  << 20));   // 4 MB
    __bf16* K  = (__bf16*)(ws + ((size_t)6  << 20));   // 4 MB
    __bf16* Vt = (__bf16*)(ws + ((size_t)10 << 20));   // 4 MB

    repack_w_kernel<<<3 * 8 * 64, 32, 0, stream>>>(Wq, Wk, Wv, wf);
    proj_qkv_kernel<<<(BATCH * SEQ) / 16, 128, 0, stream>>>(x, wf, Q, K, Vt);
    dim3 g2(SEQ / 64, BATCH);
    flash_attn_kernel<<<g2, 128, 0, stream>>>(Q, K, Vt, (float*)d_out);
}
